// Social_Aggregator_19112604467373
// MI455X (gfx1250) — matrix operand order repaired and checked by reading, not compiled, not run on hardware
//
#include <hip/hip_runtime.h>
#include <stdint.h>

typedef __attribute__((ext_vector_type(16))) _Float16 v16h;
typedef __attribute__((ext_vector_type(8)))  _Float16 v8h;
typedef __attribute__((ext_vector_type(8)))  float    v8f;
typedef __attribute__((ext_vector_type(4)))  unsigned u32x4;
typedef __attribute__((ext_vector_type(8)))  int      i32x8;
typedef __attribute__((ext_vector_type(4)))  int      i32x4;

#define D128 128

// ---- A-fragment builders, CDNA5 16-bit A layout ----------------------------
// lane holds row m=lane&15; halves 0..7 = K[kh*8..+7], 8..15 = K[16+kh*8..+7]

__device__ __forceinline__ v16h cvt16(float4 a, float4 b, float4 c, float4 d) {
  v16h r;
  r[0]=(_Float16)a.x;  r[1]=(_Float16)a.y;  r[2]=(_Float16)a.z;  r[3]=(_Float16)a.w;
  r[4]=(_Float16)b.x;  r[5]=(_Float16)b.y;  r[6]=(_Float16)b.z;  r[7]=(_Float16)b.w;
  r[8]=(_Float16)c.x;  r[9]=(_Float16)c.y;  r[10]=(_Float16)c.z; r[11]=(_Float16)c.w;
  r[12]=(_Float16)d.x; r[13]=(_Float16)d.y; r[14]=(_Float16)d.z; r[15]=(_Float16)d.w;
  return r;
}

// f32 source (fallback path): 4x b128 load + 8x v_cvt_pk
__device__ __forceinline__ v16h load_a_frag_f32(const float* __restrict__ row, int kh) {
  const float4* q0 = (const float4*)(row + kh * 8);
  const float4* q1 = (const float4*)(row + 16 + kh * 8);
  return cvt16(q0[0], q0[1], q1[0], q1[1]);
}

// f16 source (fast path / LDS H1): 2x 16B loads, zero cvt
__device__ __forceinline__ v16h load_a_frag_h(const _Float16* row, int kh) {
  v8h x0 = *(const v8h*)(row + kh * 8);
  v8h x1 = *(const v8h*)(row + 16 + kh * 8);
  v16h r;
#pragma unroll
  for (int j = 0; j < 8; ++j) { r[j] = x0[j]; r[8 + j] = x1[j]; }
  return r;
}

// B fragment from LDS f16: halves 0..7 = K[kh*8..+7], 8..15 = K[16+kh*8..+7] (2x 16B)
__device__ __forceinline__ v16h load_b_frag_lds(const _Float16* p, int kh) {
  v8h x0 = *(const v8h*)(p + kh * 8);
  v8h x1 = *(const v8h*)(p + 16 + kh * 8);
  v16h r;
#pragma unroll
  for (int j = 0; j < 8; ++j) { r[j] = x0[j]; r[8 + j] = x1[j]; }
  return r;
}

// ---- TDM: stage a flat f16 array into LDS as a 1-row 2D tensor -------------
#if __has_builtin(__builtin_amdgcn_tensor_load_to_lds)
#define HAVE_TDM 1
__device__ __forceinline__ void tdm_load_flat(unsigned lds_addr,
                                              const void* gptr,
                                              unsigned nelem /* f16 elems */) {
  unsigned long long ga = (unsigned long long)(uintptr_t)gptr;
  u32x4 g0;
  g0[0] = 1u;                                   // count=1, user descriptor
  g0[1] = lds_addr;                             // lds byte address
  g0[2] = (unsigned)(ga & 0xffffffffu);         // global_addr[31:0]
  g0[3] = (unsigned)((ga >> 32) & 0x01ffffffu)  // global_addr[56:32]
          | (2u << 30);                         // type = 2 ("image")
  i32x8 g1;
  g1[0] = (int)(1u << 16);                      // data_size = 1 (2 bytes)
  g1[1] = (int)((nelem & 0xffffu) << 16);       // tensor_dim0[15:0]
  g1[2] = (int)(((nelem >> 16) & 0xffffu)       // tensor_dim0[31:16]
          | (1u << 16));                        // tensor_dim1 = 1
  g1[3] = (int)((nelem & 0xffffu) << 16);       // tile_dim0 = nelem (<=32768)
  g1[4] = 1;                                    // tile_dim1 = 1, tile_dim2 = 0
  g1[5] = (int)nelem;                           // tensor_dim0_stride[31:0]
  g1[6] = 0;
  g1[7] = 0;
  i32x4 z4 = {0, 0, 0, 0};
  i32x8 z8 = {0, 0, 0, 0, 0, 0, 0, 0};
  __builtin_amdgcn_tensor_load_to_lds(g0, g1, z4, z4, z8, 0);
}
#else
#define HAVE_TDM 0
#endif

// ------------- Kernel 0a: weights f32 -> f16 into scratch -------------------

__global__ __launch_bounds__(256) void k_cvt_weights(
    const float* __restrict__ w1, const float* __restrict__ w2,
    _Float16* __restrict__ w1h, _Float16* __restrict__ w2h) {
  const int i = blockIdx.x * 256 + threadIdx.x;
  if (i < 128 * 256) w1h[i] = (_Float16)w1[i];
  if (i < 128 * 128) w2h[i] = (_Float16)w2[i];
}

// ------------- Kernel 0b: embedding table f32 -> f16 into scratch -----------

__global__ __launch_bounds__(256) void k_cvt_emb(
    const float* __restrict__ u2e, _Float16* __restrict__ u2eh, int n4) {
  const int i = blockIdx.x * 256 + threadIdx.x;   // one float4 per thread
  if (i < n4) {
    float4 v = ((const float4*)u2e)[i];
    v8h* dst = (v8h*)u2eh;                        // write 4 halves via half of v8h
    _Float16* p = u2eh + (size_t)i * 4;
    p[0] = (_Float16)v.x; p[1] = (_Float16)v.y;
    p[2] = (_Float16)v.z; p[3] = (_Float16)v.w;
    (void)dst;
  }
}

// ---------------- Kernel 1: per-edge MLP -> logits (WMMA) -------------------
// block = 256 threads = 8 waves; each wave = one 16-edge tile -> 128 edges/block

template <bool EMB_F16>
__global__ __launch_bounds__(256) void k_edge_mlp(
    const int* __restrict__ nodes, const int* __restrict__ neigh,
    const int* __restrict__ seg, const float* __restrict__ u2e,
    const _Float16* __restrict__ u2eh,
    const _Float16* __restrict__ w1h, const _Float16* __restrict__ w2h,
    const float* __restrict__ b1, const float* __restrict__ b2,
    const float* __restrict__ w3, const float* __restrict__ b3,
    float* __restrict__ logits, int E) {
  __shared__ _Float16 wlds1[128 * 256];      // 64 KB  W1 (f16)
  __shared__ _Float16 wlds2[128 * 128];      // 32 KB  W2 (f16)
  __shared__ _Float16 h1lds[8][16 * D128];   // 32 KB  H1 tiles (4 KB/wave)

  // ---- stage weights into LDS (TDM if available, else cooperative copy) ---
#if HAVE_TDM
  if (threadIdx.x < 32) {                    // wave 0 issues the DMA
    tdm_load_flat((unsigned)(uintptr_t)&wlds1[0], w1h, 128 * 256);
    tdm_load_flat((unsigned)(uintptr_t)&wlds2[0], w2h, 128 * 128);
    __builtin_amdgcn_s_wait_tensorcnt(0);
  }
#else
  for (int i = threadIdx.x; i < 128 * 256 / 8; i += 256)
    ((v8h*)wlds1)[i] = ((const v8h*)w1h)[i];
  for (int i = threadIdx.x; i < 128 * 128 / 8; i += 256)
    ((v8h*)wlds2)[i] = ((const v8h*)w2h)[i];
#endif
  __syncthreads();

  const int wave = threadIdx.x >> 5;
  const int lane = threadIdx.x & 31;
  const int n0   = lane & 15;                // row m (A) / col n (B,C,D)
  const int kh   = lane >> 4;
  const int tile = blockIdx.x * 8 + wave;

  int edge = tile * 16 + n0;
  if (edge >= E) edge = E - 1;               // clamp tail (writes guarded)
  const size_t rE = (size_t)neigh[edge] * D128;       // e_u row
  const size_t rU = (size_t)nodes[seg[edge]] * D128;  // u_rep row

  // ---- GEMM1: H1[16,128] = relu( X[16,256] * W1^T + b1 ) ------------------
  v8f acc1[8] = {};
  for (int ks = 0; ks < 8; ++ks) {
    const int kb = ks * 32;                  // 32-col chunk never straddles concat
    v16h a;
    if constexpr (EMB_F16) {
      const _Float16* src = (kb < D128) ? (u2eh + rE + kb) : (u2eh + rU + kb - D128);
      a = load_a_frag_h(src, kh);            // pure f16 b128 loads, no cvt
    } else {
      const float* src = (kb < D128) ? (u2e + rE + kb) : (u2e + rU + kb - D128);
      a = load_a_frag_f32(src, kh);
    }
#pragma unroll
    for (int nt = 0; nt < 8; ++nt) {
      v16h bf = load_b_frag_lds(wlds1 + (nt * 16 + n0) * 256 + kb, kh);
      acc1[nt] = __builtin_amdgcn_wmma_f32_16x16x32_f16(
          false, a, false, bf, (short)0, acc1[nt], false, false);
    }
  }

  // bias + relu, spill H1 tile to LDS as f16 (wave-private region)
  _Float16* myh1 = &h1lds[wave][0];
#pragma unroll
  for (int nt = 0; nt < 8; ++nt) {
    const int n = nt * 16 + n0;
    const float bb = b1[n];
#pragma unroll
    for (int r = 0; r < 8; ++r) {
      float v = acc1[nt][r] + bb;
      v = v > 0.f ? v : 0.f;
      myh1[(r + 8 * kh) * D128 + n] = (_Float16)v;   // D layout: m = r + 8*kh
    }
  }
  asm volatile("s_wait_dscnt 0x0" ::: "memory");     // within-wave LDS RAW fence

  // ---- GEMM2: H2[16,128] = relu( H1 * W2^T + b2 ) --------------------------
  v8f acc2[8] = {};
  for (int ks = 0; ks < 4; ++ks) {
    const int kb = ks * 32;
    v16h a = load_a_frag_h(myh1 + n0 * D128 + kb, kh);
#pragma unroll
    for (int nt = 0; nt < 8; ++nt) {
      v16h bf = load_b_frag_lds(wlds2 + (nt * 16 + n0) * 128 + kb, kh);
      acc2[nt] = __builtin_amdgcn_wmma_f32_16x16x32_f16(
          false, a, false, bf, (short)0, acc2[nt], false, false);
    }
  }

  // ---- head: logit[m] = sum_n relu(h2[m,n]+b2[n]) * w3[n]  (+ b3) ----------
  float p[8] = {0, 0, 0, 0, 0, 0, 0, 0};
#pragma unroll
  for (int nt = 0; nt < 8; ++nt) {
    const int n = nt * 16 + n0;
    const float bb = b2[n];
    const float wv = w3[n];
#pragma unroll
    for (int r = 0; r < 8; ++r) {
      float v = acc2[nt][r] + bb;
      v = v > 0.f ? v : 0.f;
      p[r] += v * wv;
    }
  }
#pragma unroll
  for (int mask = 8; mask > 0; mask >>= 1) {
#pragma unroll
    for (int r = 0; r < 8; ++r) p[r] += __shfl_xor(p[r], mask, 16);
  }
  if (n0 == 0) {
    const float bias3 = b3[0];
#pragma unroll
    for (int r = 0; r < 8; ++r) {
      const int eo = tile * 16 + r + 8 * kh;           // m = r + 8*kh
      if (eo < E) logits[eo] = p[r] + bias3;
    }
  }
}

// ------------- Kernel 2: segmented softmax over sorted seg_ids --------------

__device__ __forceinline__ int lower_bound(const int* __restrict__ a, int n, int key) {
  int lo = 0, hi = n;
  while (lo < hi) { int mid = (lo + hi) >> 1; if (a[mid] < key) lo = mid + 1; else hi = mid; }
  return lo;
}

__global__ __launch_bounds__(128) void k_softmax(
    const int* __restrict__ seg, const float* __restrict__ logits,
    float* __restrict__ att, int E) {
  __shared__ int rng[2];
  __shared__ float red[128];
  const int b = blockIdx.x, tid = threadIdx.x;
  if (tid == 0) { rng[0] = lower_bound(seg, E, b); rng[1] = lower_bound(seg, E, b + 1); }
  __syncthreads();
  const int s = rng[0], e = rng[1];

  float mx = -3.4e38f;
  for (int j = s + tid; j < e; j += 128) mx = fmaxf(mx, logits[j]);
  red[tid] = mx; __syncthreads();
  for (int off = 64; off > 0; off >>= 1) {
    if (tid < off) red[tid] = fmaxf(red[tid], red[tid + off]);
    __syncthreads();
  }
  mx = red[0]; __syncthreads();

  float sm = 0.f;
  for (int j = s + tid; j < e; j += 128) sm += __expf(logits[j] - mx);
  red[tid] = sm; __syncthreads();
  for (int off = 64; off > 0; off >>= 1) {
    if (tid < off) red[tid] += red[tid + off];
    __syncthreads();
  }
  const float inv = 1.f / red[0];
  for (int j = s + tid; j < e; j += 128) att[j] = __expf(logits[j] - mx) * inv;
}

// ------------- Kernel 3: attention-weighted aggregation ---------------------

__global__ __launch_bounds__(128) void k_aggregate(
    const int* __restrict__ seg, const int* __restrict__ neigh,
    const float* __restrict__ u2e, const float* __restrict__ att,
    float* __restrict__ out, int E) {
  __shared__ int rng[2];
  const int b = blockIdx.x, d = threadIdx.x;
  if (d == 0) { rng[0] = lower_bound(seg, E, b); rng[1] = lower_bound(seg, E, b + 1); }
  __syncthreads();
  float acc = 0.f;
  for (int j = rng[0]; j < rng[1]; ++j)
    acc += att[j] * u2e[(size_t)neigh[j] * D128 + d];   // coalesced over d
  out[(size_t)b * D128 + d] = acc;
}

// ---------------------------------------------------------------------------

extern "C" void kernel_launch(void* const* d_in, const int* in_sizes, int n_in,
                              void* d_out, int out_size, void* d_ws, size_t ws_size,
                              hipStream_t stream) {
  const int*   nodes = (const int*)d_in[0];
  const int*   neigh = (const int*)d_in[1];
  const int*   seg   = (const int*)d_in[2];
  const float* u2e   = (const float*)d_in[3];
  const float* w1    = (const float*)d_in[4];
  const float* b1    = (const float*)d_in[5];
  const float* w2    = (const float*)d_in[6];
  const float* b2    = (const float*)d_in[7];
  const float* w3    = (const float*)d_in[8];
  const float* b3    = (const float*)d_in[9];

  const int    B  = in_sizes[0];
  const int    E  = in_sizes[1];
  const size_t nU = (size_t)in_sizes[3];         // N_USERS * 128 floats

  const int grid1 = (E + 127) / 128;             // 8 waves/block, 16 edges/wave

  // fast path needs: u2e f16 (2*nU) + weights f16 (96KB) + logits/att (8*E)
  const size_t needFast = 2 * nU + 2 * (128 * 256 + 128 * 128) + (size_t)E * 8;

  if (ws_size >= needFast) {
    _Float16* u2eh   = (_Float16*)d_ws;                    // nU halves
    _Float16* w1h    = u2eh + nU;                          // 32768 halves
    _Float16* w2h    = w1h + 128 * 256;                    // 16384 halves
    float*    logits = (float*)(w2h + 128 * 128);          // E f32 (4B-aligned)
    float*    att    = logits + E;                         // E f32

    const int n4 = (int)(nU / 4);
    k_cvt_emb<<<(n4 + 255) / 256, 256, 0, stream>>>(u2e, u2eh, n4);
    k_cvt_weights<<<(128 * 256 + 255) / 256, 256, 0, stream>>>(w1, w2, w1h, w2h);
    k_edge_mlp<true><<<grid1, 256, 0, stream>>>(nodes, neigh, seg, u2e, u2eh,
                                                w1h, w2h, b1, b2, w3, b3, logits, E);
    k_softmax<<<B, 128, 0, stream>>>(seg, logits, att, E);
    k_aggregate<<<B, 128, 0, stream>>>(seg, neigh, u2e, att, (float*)d_out, E);
  } else {
    float*    logits = (float*)d_ws;
    float*    att    = logits + E;
    _Float16* w1h    = (_Float16*)(att + E);
    _Float16* w2h    = w1h + 128 * 256;

    k_cvt_weights<<<(128 * 256 + 255) / 256, 256, 0, stream>>>(w1, w2, w1h, w2h);
    k_edge_mlp<false><<<grid1, 256, 0, stream>>>(nodes, neigh, seg, u2e, nullptr,
                                                 w1h, w2h, b1, b2, w3, b3, logits, E);
    k_softmax<<<B, 128, 0, stream>>>(seg, logits, att, E);
    k_aggregate<<<B, 128, 0, stream>>>(seg, neigh, u2e, att, (float*)d_out, E);
  }
}
